// FreqMoE_40613210751004
// MI455X (gfx1250) — matrix operand order repaired and checked by reading, not compiled, not run on hardware
//
#include <hip/hip_runtime.h>
#include <hip/hip_bf16.h>

// ---------------------------------------------------------------------------
// FreqMoE FNO pipeline for gfx1250 (MI455X).
// All dense linear algebra (DFTs-as-GEMMs, spectral mixes, 1x1 conv, fc1)
// runs on v_wmma_f32_16x16x32_f16.  Twiddle bases are precomputed into
// K-padded tables (zeros in the pad) so GEMM inner loops are pure
// b128-load + wmma with no branches and no transcendentals.
// ---------------------------------------------------------------------------

typedef __attribute__((ext_vector_type(16))) _Float16 v16h;
typedef __attribute__((ext_vector_type(8)))  _Float16 v8h;
typedef __attribute__((ext_vector_type(8)))  float    v8f;
typedef __attribute__((ext_vector_type(4)))  float    v4f;

#define B_   16
#define T_   10
#define C_   64
#define HP   130          // padded spatial (128+2)
#define KP   160          // 130 padded to multiple of 32
#define NW   66           // rfft width
#define MD   12           // retained modes per dim
#define E_   15
#define R_   4
#define HW2  (HP*HP)                  // 16900
#define NF_H ((size_t)B_*C_*HW2)      // 17,305,600
#define NROW (B_*C_*HP)               // 133,120
#define PHI  (6.283185307179586f/130.0f)
#define VZ8  ((v8f){0.f,0.f,0.f,0.f,0.f,0.f,0.f,0.f})

__device__ __forceinline__ float geluf(float x){
  return 0.5f*x*(1.0f+erff(x*0.7071067811865475f));
}

// ---- fragment loads (contiguous k order, shared by A and B) ---------------------------------
// A (16x32 f16): lane L -> row M=L&15; element j -> K = ((L&16)?16:0) + j
//   => one contiguous 16-half run at [k0+hA .. +15]
// B (32x16 f16): lane L -> col N=L&15; element j -> K = ((L&16)?16:0) + j
//   => one contiguous 16-half run at [k0+hB .. +15]
__device__ __forceinline__ v16h frag_ld_a(const _Float16* __restrict__ p){
  const v8h lo = *(const v8h*)(p);
  const v8h hi = *(const v8h*)(p + 8);
  return __builtin_shufflevector(lo,hi,0,1,2,3,4,5,6,7,8,9,10,11,12,13,14,15);
}
__device__ __forceinline__ v16h frag_ld_b(const _Float16* __restrict__ p){
  const v8h lo = *(const v8h*)(p);
  const v8h hi = *(const v8h*)(p + 8);
  return __builtin_shufflevector(lo,hi,0,1,2,3,4,5,6,7,8,9,10,11,12,13,14,15);
}
__device__ __forceinline__ v16h frag_ld_b_f32(const float* __restrict__ p){
  v16h r;
#pragma unroll
  for(int q=0;q<4;++q){
    const v4f t = *(const v4f*)(p + 4*q);
#pragma unroll
    for(int e=0;e<4;++e) r[4*q+e] = (_Float16)t[e];
  }
  return r;
}
__device__ __forceinline__ v16h hneg(v16h x){
#pragma unroll
  for(int j=0;j<16;++j) x[j] = (_Float16)(-(float)x[j]);
  return x;
}
__device__ __forceinline__ v8f wmma(v16h a, v16h b, v8f c){
  return __builtin_amdgcn_wmma_f32_16x16x32_f16(false,a,false,b,(short)0,c,false,false);
}

// ---------------------------------------------------------------------------
// small utility kernels
__global__ void k_zero(float* p, int n){
  for(int i = blockIdx.x*blockDim.x + threadIdx.x; i<n; i += gridDim.x*blockDim.x) p[i]=0.f;
}
__global__ void k_zero16(_Float16* p, int n){
  for(int i = blockIdx.x*blockDim.x + threadIdx.x; i<n; i += gridDim.x*blockDim.x) p[i]=(_Float16)0.f;
}

// ---- twiddle / weight table builders (cheap, once per call / per layer) ---
__global__ void k_b_dftw(_Float16* __restrict__ TwT){      // [n<144][k<160]
  const int idx = blockIdx.x*blockDim.x+threadIdx.x; if(idx>=144*KP) return;
  const int n = idx/KP, k = idx%KP; float v = 0.f;
  if(k < HP){
    if(n < NW)        v =  __cosf(PHI*(float)((k*n)%130));
    else if(n < 2*NW) v = -__sinf(PHI*(float)((k*(n-NW))%130));
  }
  TwT[idx] = (_Float16)v;
}
__global__ void k_b_dfth(_Float16* __restrict__ TcH, _Float16* __restrict__ TsH){ // [kx<144][k<160]
  const int idx = blockIdx.x*blockDim.x+threadIdx.x; if(idx>=144*KP) return;
  const int kx = idx/KP, k = idx%KP; float c=0.f, s=0.f;
  if(k < HP){ const int ph=(kx*k)%130; c=__cosf(PHI*ph); s=-__sinf(PHI*ph); } // e^{-i t}
  TcH[idx]=(_Float16)c; TsH[idx]=(_Float16)s;
}
__global__ void k_b_ih(_Float16* __restrict__ FihC, _Float16* __restrict__ FihS){ // [h<144][j<32]
  const int idx = blockIdx.x*blockDim.x+threadIdx.x; if(idx>=144*32) return;
  const int h = idx/32, j = idx%32; float c=0.f, s=0.f;
  if(j < 24){
    const int kxx = (j<MD)? j : (HP-24)+j;
    const int ph = (h*kxx)%130;
    c = __cosf(PHI*ph)*(1.0f/130.0f); s = __sinf(PHI*ph)*(1.0f/130.0f);   // e^{+i t}/130
  }
  FihC[idx]=(_Float16)c; FihS[idx]=(_Float16)s;
}
__global__ void k_b_iw(_Float16* __restrict__ BiwT){       // [w<144][j<32]
  const int idx = blockIdx.x*blockDim.x+threadIdx.x; if(idx>=144*32) return;
  const int w = idx/32, j = idx%32; float v = 0.f;
  if(j < MD){
    const float sc = (j==0)?1.f:2.f;
    v =  sc*__cosf(PHI*(float)((j*w)%130))*(1.0f/130.0f);
  }else if(j < 24){
    const int kk = j-MD; const float sc = (kk==0)?0.f:2.f;
    v = -sc*__sinf(PHI*(float)((kk*w)%130))*(1.0f/130.0f);
  }
  BiwT[idx] = (_Float16)v;
}
__global__ void k_b_fc1(const float* __restrict__ w1, _Float16* __restrict__ fc1T){ // [n<128][k<64]
  const int idx = blockIdx.x*blockDim.x+threadIdx.x; if(idx>=128*64) return;
  fc1T[idx] = (_Float16)w1[(idx&63)*128 + (idx>>6)];
}
__global__ void k_b_wt(const float* __restrict__ Wa, const float* __restrict__ Wb,
                       _Float16* __restrict__ WTre, _Float16* __restrict__ WTim){ // [half][xy][o][i]
  const int idx = blockIdx.x*blockDim.x+threadIdx.x; if(idx>=2*144*64*64) return;
  const int i = idx & 63, o = (idx>>6)&63, q = idx>>12;
  const int half = q/144, xy = q%144, xm = xy/MD, ym = xy%MD;
  const float* w = half ? Wb : Wa;
  const size_t s = ((((size_t)i*C_+o)*MD+xm)*MD+ym)*2;
  WTre[idx]=(_Float16)w[s]; WTim[idx]=(_Float16)w[s+1];
}
__global__ void k_cvt_cw(const float* __restrict__ cw, _Float16* __restrict__ cwT){ // [o][i]
  const int idx = blockIdx.x*blockDim.x+threadIdx.x; if(idx>=C_*C_) return;
  cwT[idx] = (_Float16)cw[idx];
}

// ---- activation conversions ----------------------------------------------
// channel-major, K-padded rows: h16p[(bc*HP+h)*160 + w], zeros for w>=130
__global__ void k_cvt_pad(const float* __restrict__ h, _Float16* __restrict__ h16p){
  const int n = NROW*KP;
  for(int i = blockIdx.x*blockDim.x + threadIdx.x; i<n; i += gridDim.x*blockDim.x){
    const int row = i/KP, w = i - row*KP;
    __builtin_prefetch(h + (size_t)row*HP + w + 4096, 0, 0);
    h16p[i] = (w < HP) ? (_Float16)h[(size_t)row*HP + w] : (_Float16)0.f;
  }
}
// pixel-major: hT16[(b*HW2+hw)*64 + c]
__global__ void k_cvt_T(const float* __restrict__ h, _Float16* __restrict__ hT){
  const int n = (int)NF_H;
  for(int i = blockIdx.x*blockDim.x + threadIdx.x; i<n; i += gridDim.x*blockDim.x){
    const int c = i & 63, p = i >> 6;
    const int b = p/HW2, hw = p - b*HW2;
    hT[i] = (_Float16)h[((size_t)(b*C_+c))*HW2 + hw];
  }
}

// fc0: per pixel 12->64 into padded channel-major activation
__global__ void k_fc0(const float* __restrict__ x, const float* __restrict__ grid,
                      const float* __restrict__ w, const float* __restrict__ bias,
                      float* __restrict__ h){
  __shared__ float sv[12];
  const int pix = blockIdx.x, c = threadIdx.x;
  if(c < T_)        sv[c] = x[(size_t)pix*T_ + c];
  else if(c < T_+2) sv[c] = grid[(size_t)pix*2 + (c-T_)];
  __syncthreads();
  float acc = bias[c];
#pragma unroll
  for(int t=0;t<12;++t) acc += sv[t]*w[t*C_ + c];
  const int b = pix >> 14, r = pix & 16383, hh = r >> 7, ww = r & 127;
  h[((size_t)(b*C_+c)*HP + hh)*HP + ww] = acc;
}

// ---- stage 1: real DFT along W  (133,120 x 160 x 132) ---------------------
// out transposed: s1{re,im}T[(bc*NW+ky)*160 + h]
__global__ void k_dft_w(const _Float16* __restrict__ h16p, const _Float16* __restrict__ TwT,
                        _Float16* __restrict__ s1reT, _Float16* __restrict__ s1imT){
  const int lane = threadIdx.x & 31, wv = threadIdx.x >> 5;
  const int mb = (blockIdx.x*4 + wv)*16;            // 2080*4 = 8320 tiles exact
  const int nb = blockIdx.y*16;                     // 9 tiles (pad 144)
  const int hA = (lane&16)?16:0, hB = (lane&16)?16:0;
  const _Float16* pa = h16p + (size_t)(mb + (lane&15))*KP + hA;
  const _Float16* pb = TwT  + (size_t)(nb + (lane&15))*KP + hB;
  v8f acc = VZ8;
#pragma unroll
  for(int k0=0;k0<KP;k0+=32) acc = wmma(frag_ld_a(pa+k0), frag_ld_b(pb+k0), acc);
  const int n = nb + (lane & 15);
#pragma unroll
  for(int r=0;r<8;++r){
    const int m = mb + r + ((lane&16)?8:0);
    const int bc = m/HP, h = m - bc*HP;
    if(n < NW)        s1reT[((size_t)bc*NW + n)*KP + h]      = (_Float16)acc[r];
    else if(n < 2*NW) s1imT[((size_t)bc*NW + (n-NW))*KP + h] = (_Float16)acc[r];
  }
}

// ---- stage 2: complex 130-pt DFT along H, fused gate-|ft| + mode extract --
// modes transposed for the mixes: mT{re,im}[((srow*12+ky)*16 + b)*64 + ch]
__global__ void k_dft_h(const _Float16* __restrict__ TcH, const _Float16* __restrict__ TsH,
                        const _Float16* __restrict__ s1reT, const _Float16* __restrict__ s1imT,
                        float* __restrict__ gacc,
                        _Float16* __restrict__ mTre, _Float16* __restrict__ mTim){
  const int lane = threadIdx.x & 31, wv = threadIdx.x >> 5;
  const int mb = blockIdx.x*16;                     // kx tiles (9, pad 144)
  const int nb = (blockIdx.y*4 + wv)*16;            // 66*4 = 264 tiles exact
  const int b  = blockIdx.z;
  const int nl = nb + (lane&15), ch = nl/NW, ky = nl - ch*NW;
  const int hA = (lane&16)?16:0, hB = (lane&16)?16:0;
  const _Float16* pac = TcH + (size_t)(mb + (lane&15))*KP + hA;
  const _Float16* pas = TsH + (size_t)(mb + (lane&15))*KP + hA;
  const _Float16* pbr = s1reT + ((size_t)(b*C_+ch)*NW + ky)*KP + hB;
  const _Float16* pbi = s1imT + ((size_t)(b*C_+ch)*NW + ky)*KP + hB;
  v8f aR = VZ8, aI = VZ8;
#pragma unroll
  for(int k0=0;k0<KP;k0+=32){
    const v16h Ar = frag_ld_a(pac+k0), Ai = frag_ld_a(pas+k0);
    const v16h Br = frag_ld_b(pbr+k0), Bi = frag_ld_b(pbi+k0);
    aR = wmma(Ar,Br,aR); aR = wmma(hneg(Ai),Bi,aR);
    aI = wmma(Ar,Bi,aI); aI = wmma(Ai,Br,aI);
  }
#pragma unroll
  for(int r=0;r<8;++r){
    const int kx = mb + r + ((lane&16)?8:0);
    if(kx >= HP) continue;
    const float re = aR[r], im = aI[r];
    atomicAdd(&gacc[b*C_+ch], sqrtf(re*re + im*im));
    if(ky < MD){
      int srow = -1;
      if(kx < MD) srow = kx;
      else if(kx >= HP-MD) srow = MD + (kx-(HP-MD));
      if(srow >= 0){
        const size_t idx = ((size_t)(srow*MD+ky)*B_ + b)*C_ + ch;
        mTre[idx]=(_Float16)re; mTim[idx]=(_Float16)im;
      }
    }
  }
}

// ---- gate softmax ---------------------------------------------------------
__global__ void k_gate(const float* __restrict__ gacc, const float* __restrict__ gw,
                       const float* __restrict__ gb, float* __restrict__ gate){
  __shared__ float sl[E_];
  const int b = blockIdx.x, e = threadIdx.x;
  if(e < E_){
    float acc = gb[e];
    const float inv = 1.0f/(float)(HP*NW);
    for(int c=0;c<C_;++c) acc += gacc[b*C_+c]*inv*gw[c*E_+e];
    sl[e] = acc;
  }
  __syncthreads();
  if(e < E_){
    float m = sl[0]; for(int i=1;i<E_;++i) m = fmaxf(m, sl[i]);
    float s = 0.f;   for(int i=0;i<E_;++i) s += __expf(sl[i]-m);
    gate[b*E_+e] = __expf(sl[e]-m)/s;
  }
}

// ---- dT[b][o][i] = 0.1 * sum_e g[b,e] (A[e] @ B[e])[i,o]  (f16 output) ----
__global__ void k_dmat(const float* __restrict__ gate, const float* __restrict__ A,
                       const float* __restrict__ Bm,
                       _Float16* __restrict__ dTre, _Float16* __restrict__ dTim){
  const int idx = blockIdx.x*blockDim.x + threadIdx.x;     // (b*64+o)*64+i
  const int b = idx >> 12, o = (idx >> 6) & 63, i = idx & 63;
  float ar = 0.f, ai = 0.f;
  for(int e=0;e<E_;++e){
    const float g = gate[b*E_+e];
#pragma unroll
    for(int r=0;r<R_;++r){
      const float are = A[((e*C_+i)*R_+r)*2+0], aim = A[((e*C_+i)*R_+r)*2+1];
      const float bre = Bm[((e*R_+r)*C_+o)*2+0], bim = Bm[((e*R_+r)*C_+o)*2+1];
      ar += g*(are*bre - aim*bim);
      ai += g*(are*bim + aim*bre);
    }
  }
  dTre[idx] = (_Float16)(0.1f*ar); dTim[idx] = (_Float16)(0.1f*ai);
}

// ---- per-mode static mix: out[b,o](xy) = sum_i modes[b,i](xy) W[i,o](xy) --
// moT{re,im}[((b*64+o)*12+ym)*32 + srow]   (K-padded for the inverse DFT)
__global__ void k_mixA(const _Float16* __restrict__ mTre, const _Float16* __restrict__ mTim,
                       const _Float16* __restrict__ WTre, const _Float16* __restrict__ WTim,
                       float* __restrict__ moTre, float* __restrict__ moTim, int half){
  const int lane = threadIdx.x & 31, wv = threadIdx.x >> 5;
  const int xy = blockIdx.x, xm = xy/MD, ym = xy%MD;
  const int nb = wv*16;                             // 4 o-tiles exact
  const int srow = half*MD + xm;
  const int hA = (lane&16)?16:0, hB = (lane&16)?16:0;
  const _Float16* par = mTre + ((size_t)(srow*MD+ym)*B_ + (lane&15))*C_ + hA;
  const _Float16* pai = mTim + ((size_t)(srow*MD+ym)*B_ + (lane&15))*C_ + hA;
  const _Float16* pbr = WTre + (((size_t)(half*144+xy)*C_ + nb + (lane&15)))*C_ + hB;
  const _Float16* pbi = WTim + (((size_t)(half*144+xy)*C_ + nb + (lane&15)))*C_ + hB;
  v8f aR = VZ8, aI = VZ8;
#pragma unroll
  for(int k0=0;k0<C_;k0+=32){
    const v16h Ar = frag_ld_a(par+k0), Ai = frag_ld_a(pai+k0);
    const v16h Br = frag_ld_b(pbr+k0), Bi = frag_ld_b(pbi+k0);
    aR = wmma(Ar,Br,aR); aR = wmma(hneg(Ai),Bi,aR);
    aI = wmma(Ar,Bi,aI); aI = wmma(Ai,Br,aI);
  }
  const int o = nb + (lane & 15);
#pragma unroll
  for(int r=0;r<8;++r){
    const int b = r + ((lane&16)?8:0);
    const size_t idx = ((size_t)(b*C_+o)*MD + ym)*32 + srow;
    moTre[idx] = aR[r]; moTim[idx] = aI[r];
  }
}

// ---- per-batch MoE mix: out[b,o](xy) += sum_i modes[b,i](xy) d[b,i,o] -----
__global__ void k_mixB(const _Float16* __restrict__ mTre, const _Float16* __restrict__ mTim,
                       const _Float16* __restrict__ dTre, const _Float16* __restrict__ dTim,
                       float* __restrict__ moTre, float* __restrict__ moTim, int half){
  const int lane = threadIdx.x & 31, wv = threadIdx.x >> 5;
  const int mb = blockIdx.x*16;                     // 9 xy-tiles exact (144)
  const int nb = wv*16;                             // 4 o-tiles exact
  const int b  = blockIdx.z;
  const int xyl = mb + (lane&15), xm = xyl/MD, ym = xyl%MD;
  const int hA = (lane&16)?16:0, hB = (lane&16)?16:0;
  const _Float16* par = mTre + ((size_t)((half*MD+xm)*MD+ym)*B_ + b)*C_ + hA;
  const _Float16* pai = mTim + ((size_t)((half*MD+xm)*MD+ym)*B_ + b)*C_ + hA;
  const _Float16* pbr = dTre + ((size_t)(b*C_ + nb + (lane&15)))*C_ + hB;
  const _Float16* pbi = dTim + ((size_t)(b*C_ + nb + (lane&15)))*C_ + hB;
  v8f aR = VZ8, aI = VZ8;
#pragma unroll
  for(int k0=0;k0<C_;k0+=32){
    const v16h Ar = frag_ld_a(par+k0), Ai = frag_ld_a(pai+k0);
    const v16h Br = frag_ld_b(pbr+k0), Bi = frag_ld_b(pbi+k0);
    aR = wmma(Ar,Br,aR); aR = wmma(hneg(Ai),Bi,aR);
    aI = wmma(Ar,Bi,aI); aI = wmma(Ai,Br,aI);
  }
  const int o = nb + (lane & 15);
#pragma unroll
  for(int r=0;r<8;++r){
    const int xyr = mb + r + ((lane&16)?8:0);
    const int xr = xyr/MD, yr = xyr%MD;
    const size_t idx = ((size_t)(b*C_+o)*MD + yr)*32 + (half*MD + xr);
    moTre[idx] += aR[r]; moTim[idx] += aI[r];
  }
}

// ---- inverse stage A: complex ifft along H (24 nonzero rows, K padded 32) -
// saP[(bo*HP+h)*32 + j] : j<12 Re, 12..23 Im, 24..31 zero
__global__ void k_idft_h(const _Float16* __restrict__ FihC, const _Float16* __restrict__ FihS,
                         const float* __restrict__ moTre, const float* __restrict__ moTim,
                         _Float16* __restrict__ saP){
  const int lane = threadIdx.x & 31, wv = threadIdx.x >> 5;
  const int mb = blockIdx.x*16;                     // h tiles (9, pad 144)
  const int nb = (blockIdx.y*4 + wv)*16;            // 192*4 = 768 tiles exact
  const int hA = (lane&16)?16:0, hB = (lane&16)?16:0;
  const _Float16* pac = FihC + (size_t)(mb + (lane&15))*32 + hA;
  const _Float16* pas = FihS + (size_t)(mb + (lane&15))*32 + hA;
  const float*    pbr = moTre + (size_t)(nb + (lane&15))*32 + hB;
  const float*    pbi = moTim + (size_t)(nb + (lane&15))*32 + hB;
  v8f aR = VZ8, aI = VZ8;
  {
    const v16h Ar = frag_ld_a(pac), Ai = frag_ld_a(pas);
    const v16h Br = frag_ld_b_f32(pbr), Bi = frag_ld_b_f32(pbi);
    aR = wmma(Ar,Br,aR); aR = wmma(hneg(Ai),Bi,aR);
    aI = wmma(Ar,Bi,aI); aI = wmma(Ai,Br,aI);
  }
  const int n = nb + (lane & 15), bo = n/MD, ky = n - bo*MD;
#pragma unroll
  for(int r=0;r<8;++r){
    const int h = mb + r + ((lane&16)?8:0);
    if(h < HP){
      saP[((size_t)bo*HP + h)*32 + ky]      = (_Float16)aR[r];
      saP[((size_t)bo*HP + h)*32 + MD + ky] = (_Float16)aI[r];
    }
  }
}

// ---- inverse stage B: Hermitian real reconstruction along W ---------------
__global__ void k_idft_w(const _Float16* __restrict__ saP, const _Float16* __restrict__ BiwT,
                         float* __restrict__ s_out){
  const int lane = threadIdx.x & 31, wv = threadIdx.x >> 5;
  const int mb = (blockIdx.x*4 + wv)*16;            // 2080*4 = 8320 exact
  const int nb = blockIdx.y*16;                     // w tiles (9, pad 144)
  const int hA = (lane&16)?16:0, hB = (lane&16)?16:0;
  const _Float16* pa = saP  + (size_t)(mb + (lane&15))*32 + hA;
  const _Float16* pb = BiwT + (size_t)(nb + (lane&15))*32 + hB;
  v8f acc = wmma(frag_ld_a(pa), frag_ld_b(pb), VZ8);
  const int n = nb + (lane & 15);
  if(n < HP){
#pragma unroll
    for(int r=0;r<8;++r){
      const int row = mb + r + ((lane&16)?8:0);
      s_out[(size_t)row*HP + n] = acc[r];
    }
  }
}

// ---- 1x1 conv (M=pixels, N=o) + spectral add + bias + (gelu) -------------
__global__ void k_conv(const _Float16* __restrict__ hT16, const _Float16* __restrict__ cwT,
                       const float* __restrict__ sp, const float* __restrict__ cb,
                       float* __restrict__ hout, int do_gelu){
  const int lane = threadIdx.x & 31, wv = threadIdx.x >> 5;
  const int hwb = blockIdx.x*16;                    // 1057 tiles (guard)
  const int ob  = wv*16;                            // 4 o-tiles exact
  const int b   = blockIdx.z;
  const int hA = (lane&16)?16:0, hB = (lane&16)?16:0;
  const int hwr = min(hwb + (lane&15), HW2-1);
  const _Float16* pa = hT16 + ((size_t)b*HW2 + hwr)*C_ + hA;
  const _Float16* pb = cwT  + (size_t)(ob + (lane&15))*C_ + hB;
  v8f acc = VZ8;
#pragma unroll
  for(int k0=0;k0<C_;k0+=32) acc = wmma(frag_ld_a(pa+k0), frag_ld_b(pb+k0), acc);
  const int o = ob + (lane & 15);
  const float bias = cb[o];
#pragma unroll
  for(int r=0;r<8;++r){
    const int hw = hwb + r + ((lane&16)?8:0);
    if(hw < HW2){
      float v = acc[r] + sp[(size_t)(b*C_+o)*HW2 + hw] + bias;
      if(do_gelu) v = geluf(v);
      hout[(size_t)(b*C_+o)*HW2 + hw] = v;
    }
  }
}

__global__ void k_out_init(float* __restrict__ out, const float* __restrict__ b2, int n){
  const int i = blockIdx.x*blockDim.x + threadIdx.x;
  if(i < n) out[i] = b2[0];
}

// ---- fc1 (64->128 WMMA) + gelu + fused fc2 dot (shfl_xor reduce) ----------
__global__ void k_fc12(const _Float16* __restrict__ hT16, const _Float16* __restrict__ fc1T,
                       const float* __restrict__ b1, const float* __restrict__ w2,
                       float* __restrict__ out){
  const int lane = threadIdx.x & 31, wv = threadIdx.x >> 5;
  const int mb = blockIdx.x*16;                     // 16384 row tiles exact
  const int nb = (blockIdx.y*4 + wv)*16;            // 2*4 = 8 col tiles exact
  const int hA = (lane&16)?16:0, hB = (lane&16)?16:0;
  const int row = mb + (lane&15);
  const int b = row >> 14, rem = row & 16383, hh = rem >> 7, ww = rem & 127;
  const _Float16* pa = hT16 + ((size_t)b*HW2 + hh*HP + ww)*C_ + hA;
  const _Float16* pb = fc1T + (size_t)(nb + (lane&15))*C_ + hB;
  v8f acc = VZ8;
#pragma unroll
  for(int k0=0;k0<C_;k0+=32) acc = wmma(frag_ld_a(pa+k0), frag_ld_b(pb+k0), acc);
  const int n = nb + (lane & 15);
  const float wn = w2[n], bn = b1[n];
#pragma unroll
  for(int r=0;r<8;++r){
    float t = geluf(acc[r] + bn) * wn;
    t += __shfl_xor(t, 1, 32);
    t += __shfl_xor(t, 2, 32);
    t += __shfl_xor(t, 4, 32);
    t += __shfl_xor(t, 8, 32);
    if((lane & 15) == 0){
      const int rw = mb + r + ((lane&16)?8:0);
      atomicAdd(&out[rw], t);
    }
  }
}

// ---------------------------------------------------------------------------
extern "C" void kernel_launch(void* const* d_in, const int* in_sizes, int n_in,
                              void* d_out, int out_size, void* d_ws, size_t ws_size,
                              hipStream_t stream) {
  (void)in_sizes; (void)n_in; (void)out_size; (void)ws_size;
  const float* x      = (const float*)d_in[0];
  const float* grid   = (const float*)d_in[1];
  const float* fc0_w  = (const float*)d_in[2];
  const float* fc0_b  = (const float*)d_in[3];
  const float* W1     = (const float*)d_in[4];
  const float* W2     = (const float*)d_in[5];
  const float* A1     = (const float*)d_in[6];
  const float* B1     = (const float*)d_in[7];
  const float* A2     = (const float*)d_in[8];
  const float* B2     = (const float*)d_in[9];
  const float* gate_w = (const float*)d_in[10];
  const float* gate_b = (const float*)d_in[11];
  const float* cw     = (const float*)d_in[12];
  const float* cb     = (const float*)d_in[13];
  const float* fc1_w  = (const float*)d_in[14];
  const float* fc1_b  = (const float*)d_in[15];
  const float* fc2_w  = (const float*)d_in[16];
  const float* fc2_b  = (const float*)d_in[17];
  float* out = (float*)d_out;

  // ---- workspace (256B-aligned bump allocator) ----
  char* base = (char*)d_ws; size_t off = 0;
  auto take = [&](size_t bytes)->char*{
    char* p = base + off; off = (off + bytes + 255) & ~(size_t)255; return p; };
  float*    h_cur = (float*)   take(NF_H*4);
  _Float16* h16p  = (_Float16*)take((size_t)NROW*KP*2);
  _Float16* hT16  = (_Float16*)take(NF_H*2);
  const size_t S1N = (size_t)B_*C_*NW*KP;            // 10,813,440
  _Float16* s1reT = (_Float16*)take(S1N*2);
  _Float16* s1imT = (_Float16*)take(S1N*2);
  float*    gacc  = (float*)   take((size_t)B_*C_*4);
  float*    gate  = (float*)   take((size_t)B_*E_*4);
  const size_t MN = (size_t)24*MD*B_*C_;             // 294,912
  _Float16* mTre  = (_Float16*)take(MN*2);
  _Float16* mTim  = (_Float16*)take(MN*2);
  _Float16* d1Tre = (_Float16*)take((size_t)B_*C_*C_*2);
  _Float16* d1Tim = (_Float16*)take((size_t)B_*C_*C_*2);
  _Float16* d2Tre = (_Float16*)take((size_t)B_*C_*C_*2);
  _Float16* d2Tim = (_Float16*)take((size_t)B_*C_*C_*2);
  const size_t MON = (size_t)B_*C_*MD*32;            // 393,216
  float*    moTre = (float*)take(MON*4);
  float*    moTim = (float*)take(MON*4);
  _Float16* saP   = (_Float16*)take((size_t)NROW*32*2);
  float*    s_out = (float*)take(NF_H*4);
  _Float16* TwT   = (_Float16*)take((size_t)144*KP*2);
  _Float16* TcH   = (_Float16*)take((size_t)144*KP*2);
  _Float16* TsH   = (_Float16*)take((size_t)144*KP*2);
  _Float16* FihC  = (_Float16*)take((size_t)144*32*2);
  _Float16* FihS  = (_Float16*)take((size_t)144*32*2);
  _Float16* BiwT  = (_Float16*)take((size_t)144*32*2);
  _Float16* fc1T  = (_Float16*)take((size_t)128*C_*2);
  _Float16* WTre  = (_Float16*)take((size_t)2*144*C_*C_*2);
  _Float16* WTim  = (_Float16*)take((size_t)2*144*C_*C_*2);
  _Float16* cwT   = (_Float16*)take((size_t)C_*C_*2);

  // ---- once-per-call: tables + zero the K-pad tails -----------------------
  k_b_dftw<<<(144*KP+255)/256,256,0,stream>>>(TwT);
  k_b_dfth<<<(144*KP+255)/256,256,0,stream>>>(TcH, TsH);
  k_b_ih  <<<(144*32+255)/256,256,0,stream>>>(FihC, FihS);
  k_b_iw  <<<(144*32+255)/256,256,0,stream>>>(BiwT);
  k_b_fc1 <<<(128*64+255)/256,256,0,stream>>>(fc1_w, fc1T);
  k_zero16<<<4096,256,0,stream>>>(s1reT, (int)S1N);
  k_zero16<<<4096,256,0,stream>>>(s1imT, (int)S1N);
  k_zero  <<<1024,256,0,stream>>>(moTre, (int)MON);
  k_zero  <<<1024,256,0,stream>>>(moTim, (int)MON);
  k_zero16<<<2048,256,0,stream>>>(saP, NROW*32);

  // ---- fc0 into zero-padded activation ------------------------------------
  k_zero<<<8192,256,0,stream>>>(h_cur, (int)NF_H);
  k_fc0<<<B_*128*128,64,0,stream>>>(x, grid, fc0_w, fc0_b, h_cur);

  const size_t w_str  = (size_t)C_*C_*MD*MD*2;
  const size_t ab_str = (size_t)E_*C_*R_*2;

  for(int l=0;l<4;++l){
    const float* W1l = W1 + l*w_str;
    const float* W2l = W2 + l*w_str;
    const float* A1l = A1 + l*ab_str; const float* B1l = B1 + l*ab_str;
    const float* A2l = A2 + l*ab_str; const float* B2l = B2 + l*ab_str;
    const float* gwl = gate_w + l*C_*E_;
    const float* gbl = gate_b + l*E_;
    const float* cwl = cw + l*C_*C_;
    const float* cbl = cb + l*C_;

    k_cvt_pad<<<8192,256,0,stream>>>(h_cur, h16p);
    k_cvt_T  <<<8192,256,0,stream>>>(h_cur, hT16);
    k_zero   <<<4,256,0,stream>>>(gacc, B_*C_);
    k_cvt_cw <<<16,256,0,stream>>>(cwl, cwT);
    k_b_wt   <<<(2*144*64*64+255)/256,256,0,stream>>>(W1l, W2l, WTre, WTim);

    k_dft_w<<<dim3(NROW/64, 9),        128, 0, stream>>>(h16p, TwT, s1reT, s1imT);
    k_dft_h<<<dim3(9, 66, B_),         128, 0, stream>>>(TcH, TsH, s1reT, s1imT, gacc, mTre, mTim);

    k_gate<<<B_,32,0,stream>>>(gacc, gwl, gbl, gate);
    k_dmat<<<256,256,0,stream>>>(gate, A1l, B1l, d1Tre, d1Tim);
    k_dmat<<<256,256,0,stream>>>(gate, A2l, B2l, d2Tre, d2Tim);

    k_mixA<<<dim3(144,1),              128, 0, stream>>>(mTre, mTim, WTre, WTim, moTre, moTim, 0);
    k_mixA<<<dim3(144,1),              128, 0, stream>>>(mTre, mTim, WTre, WTim, moTre, moTim, 1);
    k_mixB<<<dim3(9,1,B_),             128, 0, stream>>>(mTre, mTim, d1Tre, d1Tim, moTre, moTim, 0);
    k_mixB<<<dim3(9,1,B_),             128, 0, stream>>>(mTre, mTim, d2Tre, d2Tim, moTre, moTim, 1);

    k_idft_h<<<dim3(9, 192),           128, 0, stream>>>(FihC, FihS, moTre, moTim, saP);
    k_idft_w<<<dim3(NROW/64, 9),       128, 0, stream>>>(saP, BiwT, s_out);

    k_conv<<<dim3((HW2+15)/16, 1, B_), 128, 0, stream>>>(
        hT16, cwT, s_out, cbl, h_cur, (l<3)?1:0);
  }

  // ---- head: crop + fc1 + gelu + fc2 (fused) ------------------------------
  k_cvt_T<<<8192,256,0,stream>>>(h_cur, hT16);
  k_out_init<<<(B_*128*128+255)/256,256,0,stream>>>(out, fc2_b, B_*128*128);
  k_fc12<<<dim3((B_*128*128)/16, 2),  128, 0, stream>>>(hT16, fc1T, fc1_b, fc2_w, out);
}
